// MinLSTM_13065290514806
// MI455X (gfx1250) — matrix operand order repaired and checked by reading, not compiled, not run on hardware
//
#include <hip/hip_runtime.h>
#include <hip/hip_bf16.h>

// ---------------------------------------------------------------------------
// MinLSTM for MI455X (gfx1250, wave32, WMMA).
//   B=8, S=4096, In=512, H=512
// Phase 1: cvt x,W to f16.
// Phase 2: fused 3-gate GEMM via v_wmma_f32_16x16x32_f16, register-blocked
//          4 M-tiles per wave, software-pipelined (double-buffered fragments),
//          fast-rcp sigmoid epilogue producing f_n and g = i_n*h_tilde (fp32)
//          in workspace.
// Phase 3: chunked parallel linear-recurrence scan (compose / carry / apply).
// ---------------------------------------------------------------------------

typedef __attribute__((ext_vector_type(16))) _Float16 v16h;
typedef __attribute__((ext_vector_type(8)))  _Float16 v8h;
typedef __attribute__((ext_vector_type(4)))  _Float16 v4h;
typedef __attribute__((ext_vector_type(8)))  float    v8f;
typedef __attribute__((ext_vector_type(4)))  float    v4f;

#define BB   8
#define SEQ  4096
#define IND  512
#define HD   512
#define NW   ((size_t)HD * IND)  // elements per weight matrix
#define NT   (HD / 16)           // 32 N-tiles
#define MT   ((BB * SEQ) / 16)   // 2048 M-tiles
#define MREP 4                   // M-tiles per wave (register blocking)
#define KSTEPS (IND / 32)        // 16 K-slices
#define CH   64                  // chunks per sequence
#define CL   (SEQ / CH)          // 64 steps per chunk
#define EPSV 1e-7f

// ---------------- fp32 -> f16 conversion (x4 vectorized) -------------------
__global__ __launch_bounds__(256) void cvt_f32_f16(const float* __restrict__ src,
                                                   _Float16* __restrict__ dst,
                                                   int n4) {
    int i = blockIdx.x * blockDim.x + threadIdx.x;
    if (i < n4) {
        v4f v = *(const v4f*)(src + (size_t)i * 4);
        v4h h;
        h[0] = (_Float16)v[0]; h[1] = (_Float16)v[1];
        h[2] = (_Float16)v[2]; h[3] = (_Float16)v[3];
        *(v4h*)(dst + (size_t)i * 4) = h;
    }
}

// ---------------- fused 3-gate WMMA GEMM + epilogue -------------------------
// xh : [B*S, IN] f16 row-major
// wh : [3, HD, IN] f16 row-major (W_f, W_i, W_h)
// Fn : [B*S, HD] f32 (normalized forget gate)
// G  : [B*S, HD] f32 (i_n * h_tilde)
__global__ __launch_bounds__(256)
void gates_wmma(const _Float16* __restrict__ xh,
                const _Float16* __restrict__ wh,
                const float* __restrict__ bfv,
                const float* __restrict__ biv,
                const float* __restrict__ bhv,
                float* __restrict__ Fn,
                float* __restrict__ G) {
    const int lane = threadIdx.x & 31;
    const int wave = threadIdx.x >> 5;
    const int wid  = blockIdx.x * 8 + wave;   // 0 .. (MT/MREP)*NT-1 (exact grid)
    const int mt   = wid / NT;                // super M-tile (4 tiles tall)
    const int nt   = wid - mt * NT;
    const int m0   = mt * (16 * MREP);
    const int n0   = nt * 16;

    const int half = lane >> 4;   // 0: K-low half of frag, 1: K-high half
    const int lid  = lane & 15;

    // A fragment source: row (m0+lid), per-lane K group kb..kb+15 (contiguous)
    const _Float16* arow = xh + (size_t)(m0 + lid) * IND + half * 16;
    const size_t rstep = (size_t)16 * IND;    // one M-tile down
    // B fragment source: W row (n0+lid) == column n of W^T; 16 contiguous halves
    const size_t woff = (size_t)(n0 + lid) * IND + half * 16;
    const _Float16* wf = wh + woff;
    const _Float16* wi = wh + NW + woff;
    const _Float16* wt = wh + 2 * NW + woff;

    v8f accF[MREP], accI[MREP], accH[MREP];
    #pragma unroll
    for (int r = 0; r < MREP; ++r) { accF[r] = v8f{}; accI[r] = v8f{}; accH[r] = v8f{}; }

    // Double-buffered fragments: loads for slice k+32 issue before the 12
    // WMMAs of slice k, so a full WMMA burst covers the load latency.
    v16h bFb[2], bIb[2], bHb[2], ab[2][MREP];

#define LOADK(buf, kk)                                                         \
    do {                                                                       \
        bFb[buf] = *(const v16h*)(wf + (kk));                                  \
        bIb[buf] = *(const v16h*)(wi + (kk));                                  \
        bHb[buf] = *(const v16h*)(wt + (kk));                                  \
        _Pragma("unroll")                                                      \
        for (int r = 0; r < MREP; ++r) {                                       \
            const _Float16* ar = arow + r * rstep + (kk);                      \
            v8h a_lo = *(const v8h*)ar;                                        \
            v8h a_hi = *(const v8h*)(ar + 8);                                  \
            ab[buf][r] = __builtin_shufflevector(a_lo, a_hi,                   \
                0, 1, 2, 3, 4, 5, 6, 7, 8, 9, 10, 11, 12, 13, 14, 15);         \
        }                                                                      \
    } while (0)

#define WMMA3(buf)                                                             \
    do {                                                                       \
        _Pragma("unroll")                                                      \
        for (int r = 0; r < MREP; ++r) {                                       \
            accF[r] = __builtin_amdgcn_wmma_f32_16x16x32_f16(false, ab[buf][r],\
                      false, bFb[buf], (short)0, accF[r], false, false);       \
            accI[r] = __builtin_amdgcn_wmma_f32_16x16x32_f16(false, ab[buf][r],\
                      false, bIb[buf], (short)0, accI[r], false, false);       \
            accH[r] = __builtin_amdgcn_wmma_f32_16x16x32_f16(false, ab[buf][r],\
                      false, bHb[buf], (short)0, accH[r], false, false);       \
        }                                                                      \
    } while (0)

    LOADK(0, 0);
    #pragma unroll
    for (int s = 0; s < KSTEPS - 1; ++s) {       // fully unrolled: s constant
        const int cur = s & 1;
        LOADK(cur ^ 1, (s + 1) * 32);
        WMMA3(cur);
    }
    WMMA3((KSTEPS - 1) & 1);                     // drain last slice

#undef LOADK
#undef WMMA3

    // Epilogue. C/D layout: VGPR j, lane L -> (M = tile + j + 8*(L>>4), N = n0+(L&15))
    // Fast-rcp sigmoid: f = rcp(1+exp(-z)); normalization via one more rcp.
    const int n = n0 + lid;
    const float bf = bfv[n];
    const float bi = biv[n];
    const float bh = bhv[n];

    #pragma unroll
    for (int r = 0; r < MREP; ++r) {
        #pragma unroll
        for (int j = 0; j < 8; ++j) {
            const int m = m0 + r * 16 + j + half * 8;
            float f  = __builtin_amdgcn_rcpf(1.0f + __expf(-(accF[r][j] + bf)));
            float i  = __builtin_amdgcn_rcpf(1.0f + __expf(-(accI[r][j] + bi)));
            float ht = accH[r][j] + bh;
            float inv = __builtin_amdgcn_rcpf(f + i + EPSV);
            size_t idx = (size_t)m * HD + n;
            Fn[idx] = f * inv;
            G[idx]  = (i * inv) * ht;
        }
    }
}

// ---------------- phase 3a: per-chunk composition (F, G) --------------------
// h_out = F * h_in + G over CL steps.  tid -> (b, c, h), consecutive h coalesced.
__global__ __launch_bounds__(256)
void scan_compose(const float* __restrict__ Fn, const float* __restrict__ G,
                  float* __restrict__ cF, float* __restrict__ cG) {
    int tid = blockIdx.x * blockDim.x + threadIdx.x;   // B*CH*HD = 262144
    int h = tid & (HD - 1);
    int c = (tid >> 9) & (CH - 1);
    int b = tid >> 15;
    size_t base = ((size_t)b * SEQ + (size_t)c * CL) * HD + h;
    float F = 1.0f, Gc = 0.0f;
    for (int t = 0; t < CL; ++t) {
        float f = Fn[base];
        float g = G[base];
        Gc = __builtin_fmaf(f, Gc, g);
        F *= f;
        base += HD;
    }
    size_t ci = ((size_t)b * CH + c) * HD + h;
    cF[ci] = F;
    cG[ci] = Gc;
}

// ---------------- phase 3b: sequential carry scan over chunks ---------------
__global__ __launch_bounds__(256)
void scan_carry(const float* __restrict__ h0,
                const float* __restrict__ cF, const float* __restrict__ cG,
                float* __restrict__ startH) {
    int tid = blockIdx.x * blockDim.x + threadIdx.x;   // B*HD = 4096
    int h = tid & (HD - 1);
    int b = tid >> 9;
    float hp = h0[(size_t)b * HD + h];
    for (int c = 0; c < CH; ++c) {
        size_t idx = ((size_t)b * CH + c) * HD + h;
        startH[idx] = hp;
        hp = __builtin_fmaf(cF[idx], hp, cG[idx]);
    }
}

// ---------------- phase 3c: apply chunk scan, write outputs -----------------
__global__ __launch_bounds__(256)
void scan_apply(const float* __restrict__ Fn, const float* __restrict__ G,
                const float* __restrict__ startH, float* __restrict__ out) {
    int tid = blockIdx.x * blockDim.x + threadIdx.x;   // B*CH*HD
    int h = tid & (HD - 1);
    int c = (tid >> 9) & (CH - 1);
    int b = tid >> 15;
    float hp = startH[((size_t)b * CH + c) * HD + h];
    size_t base = ((size_t)b * SEQ + (size_t)c * CL) * HD + h;
    for (int t = 0; t < CL; ++t) {
        hp = __builtin_fmaf(Fn[base], hp, G[base]);
        out[base] = hp;
        base += HD;
    }
}

// ---------------------------------------------------------------------------
extern "C" void kernel_launch(void* const* d_in, const int* in_sizes, int n_in,
                              void* d_out, int out_size, void* d_ws, size_t ws_size,
                              hipStream_t stream) {
    const float* x   = (const float*)d_in[0];
    const float* h0  = (const float*)d_in[1];
    const float* W_f = (const float*)d_in[2];
    const float* b_f = (const float*)d_in[3];
    const float* W_i = (const float*)d_in[4];
    const float* b_i = (const float*)d_in[5];
    const float* W_h = (const float*)d_in[6];
    const float* b_h = (const float*)d_in[7];
    float* out = (float*)d_out;

    // ---- workspace layout (bytes, 256B-aligned regions) ----
    char* ws = (char*)d_ws;
    const size_t NX = (size_t)BB * SEQ * IND;          // 16,777,216
    _Float16* xh = (_Float16*)ws;                       //  32 MB
    _Float16* wh = (_Float16*)(ws + NX * 2);            // 1.5 MB (3 matrices)
    float* Fn    = (float*)(ws + NX * 2 + 3 * NW * 2);  //  64 MB
    float* G     = Fn + NX;                             //  64 MB
    float* cF    = G + NX;                              //   1 MB
    float* cG    = cF + (size_t)BB * CH * HD;           //   1 MB
    float* stH   = cG + (size_t)BB * CH * HD;           //   1 MB

    // Phase 1: conversions
    cvt_f32_f16<<<(int)(NX / 4 / 256), 256, 0, stream>>>(x, xh, (int)(NX / 4));
    cvt_f32_f16<<<(int)(NW / 4 / 256), 256, 0, stream>>>(W_f, wh, (int)(NW / 4));
    cvt_f32_f16<<<(int)(NW / 4 / 256), 256, 0, stream>>>(W_i, wh + NW, (int)(NW / 4));
    cvt_f32_f16<<<(int)(NW / 4 / 256), 256, 0, stream>>>(W_h, wh + 2 * NW, (int)(NW / 4));

    // Phase 2: fused 3-gate WMMA GEMM ((MT/MREP)*NT waves, 8 waves/block, exact)
    gates_wmma<<<(MT / MREP * NT) / 8, 256, 0, stream>>>(xh, wh, b_f, b_i, b_h, Fn, G);

    // Phase 3: chunked linear-recurrence scan
    scan_compose<<<(BB * CH * HD) / 256, 256, 0, stream>>>(Fn, G, cF, cG);
    scan_carry<<<(BB * HD) / 256, 256, 0, stream>>>(h0, cF, cG, stH);
    scan_apply<<<(BB * CH * HD) / 256, 256, 0, stream>>>(Fn, G, stH, out);
}
